// SelfAttentionHead_35167192220219
// MI455X (gfx1250) — matrix operand order repaired and checked by reading, not compiled, not run on hardware
//
#include <hip/hip_runtime.h>

typedef unsigned short u16;
typedef unsigned int   u32;
typedef __attribute__((ext_vector_type(16))) __bf16 v16bf;
typedef __attribute__((ext_vector_type(2)))  __bf16 v2bf;
typedef __attribute__((ext_vector_type(8)))  float  v8f;

#define EMBED 1024
#define KEY   64
#define CTX   4096
#define BATCH 8
#define ROWS  (BATCH * CTX)

union Frag {
    v16bf bf;
    u32   u[8];
};

// Hardware bf16 conversion (RNE) via native __bf16 casts.
__device__ __forceinline__ u16 f2bf(float f) {
    __bf16 b = (__bf16)f;
    return __builtin_bit_cast(u16, b);
}
__device__ __forceinline__ u32 pk2(float a, float b) {
    v2bf v;
    v.x = (__bf16)a;
    v.y = (__bf16)b;
    return __builtin_bit_cast(u32, v);
}

// ---- fragment builders -----------------------------------------------------
// A-matrix 16x32 bf16 layout: lane m = lane&15, h = lane>>4;
//   VGPR0..7 hold k = 16h..16h+15 (contiguous, matching B).
__device__ __forceinline__ v16bf afrag_f32(const float* __restrict__ row, int h) {
    const float* p0 = row + 16 * h;
    const float* p1 = row + 16 * h + 8;
    float4 x0 = ((const float4*)p0)[0];
    float4 x1 = ((const float4*)p0)[1];
    float4 x2 = ((const float4*)p1)[0];
    float4 x3 = ((const float4*)p1)[1];
    Frag f;
    f.u[0] = pk2(x0.x, x0.y); f.u[1] = pk2(x0.z, x0.w);
    f.u[2] = pk2(x1.x, x1.y); f.u[3] = pk2(x1.z, x1.w);
    f.u[4] = pk2(x2.x, x2.y); f.u[5] = pk2(x2.z, x2.w);
    f.u[6] = pk2(x3.x, x3.y); f.u[7] = pk2(x3.z, x3.w);
    return f.bf;
}
__device__ __forceinline__ v16bf afrag_bf16(const u16* __restrict__ row, int h) {
    uint4 x0 = *(const uint4*)(row + 16 * h);
    uint4 x1 = *(const uint4*)(row + 16 * h + 8);
    Frag f;
    f.u[0] = x0.x; f.u[1] = x0.y; f.u[2] = x0.z; f.u[3] = x0.w;
    f.u[4] = x1.x; f.u[5] = x1.y; f.u[6] = x1.z; f.u[7] = x1.w;
    return f.bf;
}
// B-matrix 32x16 bf16 layout: lane n = lane&15; lanes 0-15 k=0..15,
// lanes 16-31 k=16..31, contiguous per lane -> p points at col-row + 16h.
__device__ __forceinline__ v16bf bfrag_bf16(const u16* __restrict__ p) {
    uint4 x0 = ((const uint4*)p)[0];
    uint4 x1 = ((const uint4*)p)[1];
    Frag f;
    f.u[0] = x0.x; f.u[1] = x0.y; f.u[2] = x0.z; f.u[3] = x0.w;
    f.u[4] = x1.x; f.u[5] = x1.y; f.u[6] = x1.z; f.u[7] = x1.w;
    return f.bf;
}

__device__ __forceinline__ v8f wmma_bf16(v16bf a, v16bf b, v8f c) {
    return __builtin_amdgcn_wmma_f32_16x16x32_bf16(false, a, false, b, (short)0, c,
                                                   false, false);
}

// ---------------------------------------------------------------------------
// Kernel 0: one-shot fp32 -> bf16 conversion of all five weight matrices.
// Layout in wbuf: kw1[64*1024] | qw1[64*1024] | vw[64*1024] | kw2[64*64] | qw2[64*64]
// ---------------------------------------------------------------------------
#define W1SZ (KEY * EMBED)
#define W2SZ (KEY * KEY)
#define WTOT (3 * W1SZ + 2 * W2SZ)

__global__ __launch_bounds__(256) void cvt_w_kernel(
    const float* __restrict__ kw1, const float* __restrict__ qw1,
    const float* __restrict__ vw,  const float* __restrict__ kw2,
    const float* __restrict__ qw2, u16* __restrict__ wbuf)
{
    const int i = blockIdx.x * 256 + threadIdx.x;
    float v;
    if (i < W1SZ)            v = kw1[i];
    else if (i < 2 * W1SZ)   v = qw1[i - W1SZ];
    else if (i < 3 * W1SZ)   v = vw[i - 2 * W1SZ];
    else if (i < 3 * W1SZ + W2SZ) v = kw2[i - 3 * W1SZ];
    else                     v = qw2[i - 3 * W1SZ - W2SZ];
    wbuf[i] = f2bf(v);
}

// ---------------------------------------------------------------------------
// Kernel 1: K = relu(relu(X W1k^T+b1k) W2k^T+b2k), Q likewise, V = X Wv^T+bv.
// Q,K -> bf16 row-major [ROWS][64];  V -> bf16 transposed [B][64][CTX].
// ---------------------------------------------------------------------------
__global__ __launch_bounds__(256) void proj_kernel(
    const float* __restrict__ X,
    const u16* __restrict__ wbuf,
    const float* __restrict__ kb1, const float* __restrict__ kb2,
    const float* __restrict__ qb1, const float* __restrict__ qb2,
    const float* __restrict__ vb,
    u16* __restrict__ Qb, u16* __restrict__ Kb, u16* __restrict__ Vt)
{
    __shared__ alignas(16) u16 sH[8][16][64];   // wave-private H1 staging
    const int w    = threadIdx.x >> 5;
    const int lane = threadIdx.x & 31;
    const int h    = lane >> 4;
    const int nlo  = lane & 15;
    const int row0 = blockIdx.x * 128 + w * 16;

    const u16* kw1b = wbuf;
    const u16* qw1b = wbuf + W1SZ;
    const u16* vwb  = wbuf + 2 * W1SZ;
    const u16* kw2b = wbuf + 3 * W1SZ;
    const u16* qw2b = wbuf + 3 * W1SZ + W2SZ;

    v8f accK[4] = {}, accQ[4] = {}, accV[4] = {};
    const float* xrow = X + (size_t)(row0 + nlo) * EMBED;

    for (int kk = 0; kk < EMBED; kk += 32) {
        v16bf a = afrag_f32(xrow + kk, h);
        const int kb = kk + 16 * h;
#pragma unroll
        for (int nt = 0; nt < 4; ++nt) {
            const int col = nt * 16 + nlo;
            accK[nt] = wmma_bf16(a, bfrag_bf16(kw1b + (size_t)col * EMBED + kb), accK[nt]);
            accQ[nt] = wmma_bf16(a, bfrag_bf16(qw1b + (size_t)col * EMBED + kb), accQ[nt]);
            accV[nt] = wmma_bf16(a, bfrag_bf16(vwb  + (size_t)col * EMBED + kb), accV[nt]);
        }
    }

    // ---- K path, second 64x64 layer ----
#pragma unroll
    for (int nt = 0; nt < 4; ++nt) {
        float bias = kb1[nt * 16 + nlo];
#pragma unroll
        for (int j = 0; j < 8; ++j) {
            float v = accK[nt][j] + bias;
            sH[w][j + 8 * h][nt * 16 + nlo] = f2bf(v > 0.f ? v : 0.f);
        }
    }
    {
        v8f acc2[4] = {};
#pragma unroll
        for (int c = 0; c < 2; ++c) {
            v16bf a = afrag_bf16(&sH[w][nlo][0] + 32 * c, h);
            const int kb = 32 * c + 16 * h;
#pragma unroll
            for (int nt = 0; nt < 4; ++nt)
                acc2[nt] = wmma_bf16(a, bfrag_bf16(kw2b + (size_t)(nt * 16 + nlo) * KEY + kb),
                                     acc2[nt]);
        }
#pragma unroll
        for (int nt = 0; nt < 4; ++nt) {
            float bias = kb2[nt * 16 + nlo];
#pragma unroll
            for (int j = 0; j < 8; ++j) {
                float v = acc2[nt][j] + bias;
                v = v > 0.f ? v : 0.f;
                Kb[(size_t)(row0 + j + 8 * h) * KEY + nt * 16 + nlo] = f2bf(v);
            }
        }
    }

    // ---- Q path, second 64x64 layer (reuse sH; per-wave DS ops are in-order) ----
#pragma unroll
    for (int nt = 0; nt < 4; ++nt) {
        float bias = qb1[nt * 16 + nlo];
#pragma unroll
        for (int j = 0; j < 8; ++j) {
            float v = accQ[nt][j] + bias;
            sH[w][j + 8 * h][nt * 16 + nlo] = f2bf(v > 0.f ? v : 0.f);
        }
    }
    {
        v8f acc2[4] = {};
#pragma unroll
        for (int c = 0; c < 2; ++c) {
            v16bf a = afrag_bf16(&sH[w][nlo][0] + 32 * c, h);
            const int kb = 32 * c + 16 * h;
#pragma unroll
            for (int nt = 0; nt < 4; ++nt)
                acc2[nt] = wmma_bf16(a, bfrag_bf16(qw2b + (size_t)(nt * 16 + nlo) * KEY + kb),
                                     acc2[nt]);
        }
#pragma unroll
        for (int nt = 0; nt < 4; ++nt) {
            float bias = qb2[nt * 16 + nlo];
#pragma unroll
            for (int j = 0; j < 8; ++j) {
                float v = acc2[nt][j] + bias;
                v = v > 0.f ? v : 0.f;
                Qb[(size_t)(row0 + j + 8 * h) * KEY + nt * 16 + nlo] = f2bf(v);
            }
        }
    }

    // ---- V store, transposed: Vt[b][n][t] ----
    const int b  = row0 / CTX;
    const int t0 = row0 % CTX;
#pragma unroll
    for (int nt = 0; nt < 4; ++nt) {
        float bias = vb[nt * 16 + nlo];
        uint4 pkv;
        pkv.x = pk2(accV[nt][0] + bias, accV[nt][1] + bias);
        pkv.y = pk2(accV[nt][2] + bias, accV[nt][3] + bias);
        pkv.z = pk2(accV[nt][4] + bias, accV[nt][5] + bias);
        pkv.w = pk2(accV[nt][6] + bias, accV[nt][7] + bias);
        *(uint4*)(Vt + (size_t)b * KEY * CTX + (size_t)(nt * 16 + nlo) * CTX
                  + t0 + 8 * h) = pkv;
    }
}

// ---------------------------------------------------------------------------
// Kernel 2: streaming (flash-style) attention with the reference's exact
// masking semantics: score/8 + (-1e10) for s<=t (fp32 absorption reproduces
// the uniform-weights last row).
// ---------------------------------------------------------------------------
__global__ __launch_bounds__(256) void attn_kernel(
    const u16* __restrict__ Qb, const u16* __restrict__ Kb,
    const u16* __restrict__ Vt, float* __restrict__ out)
{
    __shared__ alignas(16) u16 sP[8][16][32];   // wave-private P staging
    const int w    = threadIdx.x >> 5;
    const int lane = threadIdx.x & 31;
    const int h    = lane >> 4;
    const int nlo  = lane & 15;
    const int gq   = blockIdx.x * 128 + w * 16;
    const int b    = gq / CTX;
    const int t0   = gq % CTX;

    const u16* qrow = Qb + (size_t)(gq + nlo) * KEY;
    const v16bf qf0 = afrag_bf16(qrow, h);
    const v16bf qf1 = afrag_bf16(qrow + 32, h);

    v8f o[4] = {};
    float m[8], l[8];
#pragma unroll
    for (int j = 0; j < 8; ++j) { m[j] = -__builtin_inff(); l[j] = 0.f; }

    const u16* Kbase = Kb + (size_t)b * CTX * KEY;
    const u16* Vbase = Vt + (size_t)b * KEY * CTX;

    for (int s0 = 0; s0 < CTX; s0 += 32) {
        const u16* krowA = Kbase + (size_t)(s0 + nlo) * KEY;
        const u16* krowB = krowA + 16 * KEY;
        // prefetch next chunk of the K stream (global_prefetch_b8)
        if (s0 + 32 < CTX)
            __builtin_prefetch(krowA + 32 * KEY, 0, 1);

        v8f Sa = {}, Sb = {};
        Sa = wmma_bf16(qf0, bfrag_bf16(krowA + 16 * h), Sa);
        Sa = wmma_bf16(qf1, bfrag_bf16(krowA + 32 + 16 * h), Sa);
        Sb = wmma_bf16(qf0, bfrag_bf16(krowB + 16 * h), Sb);
        Sb = wmma_bf16(qf1, bfrag_bf16(krowB + 32 + 16 * h), Sb);

        float fa[8], fb[8];
        const int sca = s0 + nlo, scb = s0 + 16 + nlo;
#pragma unroll
        for (int j = 0; j < 8; ++j) {
            const int tg = t0 + j + 8 * h;
            fa[j] = Sa[j] * 0.125f + (sca <= tg ? -1e10f : 0.0f);
            fb[j] = Sb[j] * 0.125f + (scb <= tg ? -1e10f : 0.0f);
        }
#pragma unroll
        for (int j = 0; j < 8; ++j) {
            float t = fmaxf(fa[j], fb[j]);
            t = fmaxf(t, __shfl_xor(t, 1, 32));
            t = fmaxf(t, __shfl_xor(t, 2, 32));
            t = fmaxf(t, __shfl_xor(t, 4, 32));
            t = fmaxf(t, __shfl_xor(t, 8, 32));
            const float mn    = fmaxf(m[j], t);
            const float alpha = __expf(m[j] - mn);
            fa[j] = __expf(fa[j] - mn);
            fb[j] = __expf(fb[j] - mn);
            float rs = fa[j] + fb[j];
            rs += __shfl_xor(rs, 1, 32);
            rs += __shfl_xor(rs, 2, 32);
            rs += __shfl_xor(rs, 4, 32);
            rs += __shfl_xor(rs, 8, 32);
            l[j] = l[j] * alpha + rs;
            m[j] = mn;
            o[0][j] = o[0][j] * alpha;
            o[1][j] = o[1][j] * alpha;
            o[2][j] = o[2][j] * alpha;
            o[3][j] = o[3][j] * alpha;
        }
#pragma unroll
        for (int j = 0; j < 8; ++j) {
            sP[w][j + 8 * h][nlo]      = f2bf(fa[j]);
            sP[w][j + 8 * h][16 + nlo] = f2bf(fb[j]);
        }
        const v16bf pf = afrag_bf16(&sP[w][nlo][0], h);
#pragma unroll
        for (int nt = 0; nt < 4; ++nt) {
            const u16* vrow = Vbase + (size_t)(nt * 16 + nlo) * CTX + s0;
            o[nt] = wmma_bf16(pf, bfrag_bf16(vrow + 16 * h), o[nt]);
        }
    }

#pragma unroll
    for (int j = 0; j < 8; ++j) {
        const float inv = 1.0f / l[j];
        const int trow  = t0 + j + 8 * h;
        float* dst = out + ((size_t)b * CTX + trow) * KEY + nlo;
        dst[0]  = o[0][j] * inv;
        dst[16] = o[1][j] * inv;
        dst[32] = o[2][j] * inv;
        dst[48] = o[3][j] * inv;
    }
}

extern "C" void kernel_launch(void* const* d_in, const int* in_sizes, int n_in,
                              void* d_out, int out_size, void* d_ws, size_t ws_size,
                              hipStream_t stream) {
    const float* X   = (const float*)d_in[0];
    const float* kw1 = (const float*)d_in[1];
    const float* kb1 = (const float*)d_in[2];
    const float* kw2 = (const float*)d_in[3];
    const float* kb2 = (const float*)d_in[4];
    const float* qw1 = (const float*)d_in[5];
    const float* qb1 = (const float*)d_in[6];
    const float* qw2 = (const float*)d_in[7];
    const float* qb2 = (const float*)d_in[8];
    const float* vw  = (const float*)d_in[9];
    const float* vb  = (const float*)d_in[10];

    u16* wbuf = (u16*)d_ws;                      // WTOT bf16 = 400 KB
    u16* Qb   = wbuf + WTOT;                     // ROWS*64 bf16 = 4 MB
    u16* Kb   = Qb + (size_t)ROWS * KEY;         // 4 MB
    u16* Vt   = Kb + (size_t)ROWS * KEY;         // 4 MB, transposed [B][64][CTX]

    cvt_w_kernel<<<WTOT / 256, 256, 0, stream>>>(kw1, qw1, vw, kw2, qw2, wbuf);
    proj_kernel<<<ROWS / 128, 256, 0, stream>>>(X, wbuf, kb1, kb2, qb1, qb2, vb,
                                                Qb, Kb, Vt);
    attn_kernel<<<ROWS / 128, 256, 0, stream>>>(Qb, Kb, Vt, (float*)d_out);
}
